// Conv2_5D_disp_5669356836235
// MI455X (gfx1250) — matrix operand order repaired and checked by reading, not compiled, not run on hardware
//
#include <hip/hip_runtime.h>

typedef _Float16 v2h  __attribute__((ext_vector_type(2)));
typedef _Float16 v8h  __attribute__((ext_vector_type(8)));
typedef _Float16 v16h __attribute__((ext_vector_type(16)));
typedef float    v8f  __attribute__((ext_vector_type(8)));

#define IN_C   32
#define OUT_C  64
#define HH     256
#define WW     256
#define TILE_W 64
#define KTOT   864        // 3 weight sets * IN_C * 9
#define BS_STRIDE 876     // padded LDS panel column stride (halves)

// ---- pre-kernel: convert/concat weights to f16 W[64][864] in workspace ----
__global__ __launch_bounds__(256)
void pack_weights(const float* __restrict__ w0,
                  const float* __restrict__ w1,
                  const float* __restrict__ w2,
                  _Float16* __restrict__ wp)
{
    int idx = blockIdx.x * 256 + threadIdx.x;      // [0, 3*64*288)
    if (idx >= 3 * OUT_C * 288) return;
    int g  = idx / (OUT_C * 288);
    int r  = idx % (OUT_C * 288);
    int oc = r / 288;
    int k  = r % 288;
    const float* src = (g == 0) ? w0 : (g == 1) ? w1 : w2;
    wp[(size_t)oc * KTOT + g * 288 + k] = (_Float16)src[oc * 288 + k];
}

__global__ __launch_bounds__(512)
void conv25d_wmma_f16(const float* __restrict__ x,
                      const float* __restrict__ disp,
                      const float* __restrict__ fx,
                      const float* __restrict__ baseline,
                      const _Float16* __restrict__ wpk,
                      float* __restrict__ out)
{
    __shared__ float    xs[IN_C][3][68];      // x halo
    __shared__ float    dsh[3][68];           // disp halo
    __shared__ float    ms[3][9][TILE_W];     // mask table
    __shared__ _Float16 bs[TILE_W * BS_STRIDE]; // f16 masked im2col, column-major

    const int tid = threadIdx.x;
    const int bid = blockIdx.x;
    const int wtiles = WW / TILE_W;                  // 4
    const int n   = bid / (HH * wtiles);
    const int r0  = bid % (HH * wtiles);
    const int h   = r0 / wtiles;
    const int w0p = (r0 % wtiles) * TILE_W;

    // ---- phase 1: stage x and disp halos (zero pad) ----
    const float* xn = x + (size_t)n * IN_C * HH * WW;
    for (int idx = tid; idx < IN_C * 3 * 66; idx += 512) {
        int c = idx / 198, rm = idx % 198, r = rm / 66, wo = rm % 66;
        int gh = h + r - 1, gw = w0p + wo - 1;
        float v = 0.0f;
        if ((unsigned)gh < (unsigned)HH && (unsigned)gw < (unsigned)WW)
            v = xn[(c * HH + gh) * WW + gw];
        xs[c][r][wo] = v;
    }
    const float* dnp = disp + (size_t)n * HH * WW;
    for (int idx = tid; idx < 3 * 66; idx += 512) {
        int r = idx / 66, wo = idx % 66;
        int gh = h + r - 1, gw = w0p + wo - 1;
        float v = 0.0f;
        if ((unsigned)gh < (unsigned)HH && (unsigned)gw < (unsigned)WW)
            v = dnp[gh * WW + gw];
        dsh[r][wo] = v;
    }
    __syncthreads();

    // ---- phase 2: per-pixel mask table ----
    const float fxv = fx[n];
    const float bfv = baseline[n] * fxv;
    for (int idx = tid; idx < 9 * TILE_W; idx += 512) {
        int p = idx / TILE_W, px = idx % TILE_W;
        float dcol = dsh[p / 3][px + (p % 3)];
        float dcen = dsh[1][px + 1];
        bool vc = (dcen != 0.0f);
        bool vv = (dcol != 0.0f) && vc;
        float depth  = bfv / fminf(fmaxf(vv ? dcol : 0.0f, 0.01f), 256.0f);
        float center = bfv / fminf(fmaxf(vc ? dcen : 0.0f, 0.01f), 256.0f);
        float grng = 16.0f * center / fxv;           // PIXEL_SIZE*DIL*center/fx
        float hf   = 0.5f * grng;
        ms[0][p][px] = (fabsf(depth - (center + grng)) <= hf) ? 1.0f : 0.0f;
        float m1 = (fabsf(depth - center) <= hf) ? 1.0f : 0.0f;
        ms[1][p][px] = vv ? m1 : 1.0f;               // clip(m1 + 1 - valid)
        ms[2][p][px] = (fabsf(depth - (center - grng)) <= hf) ? 1.0f : 0.0f;
    }
    __syncthreads();

    // ---- phase 3: build masked f16 B panel, column-major per pixel ----
    {
        const int px = tid & 63;
        const int s  = tid >> 6;                     // 0..7: channel slice
        float msr[27];                               // masks cached in regs
        #pragma unroll
        for (int j = 0; j < 27; ++j)
            msr[j] = (&ms[0][0][0])[j * TILE_W + px];
        _Float16* colp0 = &bs[px * BS_STRIDE];
        for (int c = s; c < IN_C; c += 8) {          // 4 channels / thread
            float xv[9];
            #pragma unroll
            for (int dr = 0; dr < 3; ++dr)
                #pragma unroll
                for (int dc = 0; dc < 3; ++dc)
                    xv[dr * 3 + dc] = xs[c][dr][px + dc];
            #pragma unroll
            for (int g = 0; g < 3; ++g) {
                _Float16 hv[9];
                #pragma unroll
                for (int j = 0; j < 9; ++j)
                    hv[j] = (_Float16)(xv[j] * msr[g * 9 + j]);
                _Float16* colp = colp0 + g * 288 + c * 9;
                if ((c & 1) == 0) {                  // wave-uniform (parity of s)
                    #pragma unroll
                    for (int q = 0; q < 4; ++q) {
                        v2h t = { hv[2 * q], hv[2 * q + 1] };
                        *(v2h*)(colp + 2 * q) = t;
                    }
                    colp[8] = hv[8];
                } else {
                    colp[0] = hv[0];
                    #pragma unroll
                    for (int q = 0; q < 4; ++q) {
                        v2h t = { hv[1 + 2 * q], hv[2 + 2 * q] };
                        *(v2h*)(colp + 1 + 2 * q) = t;
                    }
                }
            }
        }
    }
    __syncthreads();

    // ---- phase 4: 27 x V_WMMA_F32_16X16X32_F16, 16x16 tile per wave ----
    const int wave = tid >> 5, lane = tid & 31;
    const int ct = wave & 3, pt = wave >> 2;
    const int m  = lane & 15, lg = lane >> 4;
    const int pxl = pt * 16 + m;

    // A fragment: lane halves = 16 consecutive K starting at K+16*lg
    const _Float16* wrow = wpk + (size_t)(ct * 16 + m) * KTOT + 16 * lg;
    // B fragment: lane halves = 16 consecutive K starting at K+16*lg, col pxl
    const _Float16* bcol = &bs[pxl * BS_STRIDE + 16 * lg];
    v8f acc = {};
    #pragma unroll
    for (int kk = 0; kk < 27; ++kk) {
        v8h alo = *(const v8h*)(wrow + 32 * kk);
        v8h ahi = *(const v8h*)(wrow + 32 * kk + 8);
        v8h blo = *(const v8h*)(bcol + 32 * kk);
        v8h bhi = *(const v8h*)(bcol + 32 * kk + 8);
        v16h a = __builtin_shufflevector(alo, ahi, 0,1,2,3,4,5,6,7,8,9,10,11,12,13,14,15);
        v16h b = __builtin_shufflevector(blo, bhi, 0,1,2,3,4,5,6,7,8,9,10,11,12,13,14,15);
        acc = __builtin_amdgcn_wmma_f32_16x16x32_f16(
            false, a, false, b, (short)0, acc, false, false);
    }

    // ---- store D: vgpr v -> oc = ct*16 + v + 8*lg, col = pxl ----
    const int oc = ct * 16 + 8 * lg;
    float* op = out + (((size_t)n * OUT_C + oc) * HH + h) * WW + w0p + pxl;
    #pragma unroll
    for (int v = 0; v < 8; ++v)
        op[(size_t)v * HH * WW] = acc[v];
}

extern "C" void kernel_launch(void* const* d_in, const int* in_sizes, int n_in,
                              void* d_out, int out_size, void* d_ws, size_t ws_size,
                              hipStream_t stream) {
    const float* x        = (const float*)d_in[0];
    const float* disp     = (const float*)d_in[1];
    const float* fx       = (const float*)d_in[2];
    const float* baseline = (const float*)d_in[3];
    const float* w0       = (const float*)d_in[4];
    const float* w1       = (const float*)d_in[5];
    const float* w2       = (const float*)d_in[6];
    float* out            = (float*)d_out;
    _Float16* wpk         = (_Float16*)d_ws;       // 3*64*288*2B = 110,592 B

    pack_weights<<<216, 256, 0, stream>>>(w0, w1, w2, wpk);
    conv25d_wmma_f16<<<2 * HH * (WW / TILE_W), 512, 0, stream>>>(
        x, disp, fx, baseline, wpk, out);
}
